// MSFormer_v1_57543971832425
// MI455X (gfx1250) — compile-verified, matrix operand order checked by reading, not run on hardware
//
#include <hip/hip_runtime.h>
#include <hip/hip_bf16.h>
#include <math.h>

// ---------------------------------------------------------------------------
// MSFormer forward for MI455X (gfx1250, wave32, WMMA).
// GEMMs: v_wmma_f32_16x16x32_f16; f32 tiles double-buffered via async DMA
// (global_load_async_to_lds_b128, ASYNCcnt-pipelined: issue tile i+1, wait
// tile i with s_wait_asynccnt 0x4, compute while DMA flies). f16 convert on
// fragment load (v_cvt_pk_f16_f32). 2 WMMA per wave per K-step.
// ---------------------------------------------------------------------------

typedef _Float16 h16v __attribute__((ext_vector_type(16)));
typedef float    f4v  __attribute__((ext_vector_type(4)));
typedef float    f8v  __attribute__((ext_vector_type(8)));

#define DD   100
#define EE   400
#define IMGX 257
#define NFT  (257 * 257)
#define NLT  (129 * 129)

// ---------------- wave32 reductions ----------------
__device__ __forceinline__ float wred_sum(float v) {
#pragma unroll
  for (int o = 16; o > 0; o >>= 1) v += __shfl_xor(v, o, 32);
  return v;
}
__device__ __forceinline__ float wred_max(float v) {
#pragma unroll
  for (int o = 16; o > 0; o >>= 1) v = fmaxf(v, __shfl_xor(v, o, 32));
  return v;
}

// ---------------- WMMA fragment load from f32 LDS tile ----------------
// 16-bit A/B 16x32 fragment striping (CDNA5 ISA):
// lane L: row = L&15, half = L>>4; halfword i -> K = (i/8)*16 + (i%8) + 8*half
// => two contiguous 8-float runs at float offsets 8*half and 16+8*half.
__device__ __forceinline__ h16v frag_from_lds_f32(const float* rowp, int hi) {
  f4v x0 = *(const f4v*)(rowp + 8 * hi);
  f4v x1 = *(const f4v*)(rowp + 8 * hi + 4);
  f4v y0 = *(const f4v*)(rowp + 16 + 8 * hi);
  f4v y1 = *(const f4v*)(rowp + 16 + 8 * hi + 4);
  h16v r;
#pragma unroll
  for (int i = 0; i < 4; ++i) {
    r[i] = (_Float16)x0[i];
    r[4 + i] = (_Float16)x1[i];
    r[8 + i] = (_Float16)y0[i];
    r[12 + i] = (_Float16)y1[i];
  }
  return r;
}

__device__ __forceinline__ unsigned lds_off_u32(const void* p) {
  return (unsigned)(unsigned long long)p;  // LDS aperture: low 32 bits = offset
}
__device__ __forceinline__ void async_b128(unsigned lds_addr, const void* gaddr) {
  asm volatile("global_load_async_to_lds_b128 %0, %1, off"
               :: "v"(lds_addr), "v"(gaddr)
               : "memory");
}
__device__ __forceinline__ void wait_async0() {
  asm volatile("s_wait_asynccnt 0x0" ::: "memory");
}
__device__ __forceinline__ void wait_async4() {
  asm volatile("s_wait_asynccnt 0x4" ::: "memory");  // drain previous tile (4 ops)
}

// ---------------------------------------------------------------------------
// GEMM: C[M,N] = alpha * A[M,K] @ Wt[N,K]^T + bias + residual
// (all [K,N]-major weights are pre-transposed on device).
// 256 threads = 8 waves (4x2); block tile 64x64; wave tile 16x32; BK = 32.
// Interior tiles: async-DMA double-buffered pipeline. Edge tiles: branchless
// clamped scalar fill.
// ---------------------------------------------------------------------------
__global__ __launch_bounds__(256) void gemm_wmma_f16(
    const float* __restrict__ A, const float* __restrict__ W,
    const float* __restrict__ bias, const float* __restrict__ residual,
    float* __restrict__ C, int M, int K, int N, float alpha) {
  __shared__ __align__(16) float sA[2][64][36];
  __shared__ __align__(16) float sB[2][64][36];
  const int t = threadIdx.x;
  const int lane = t & 31, wave = t >> 5;
  const int wm = wave & 3, wn = wave >> 2;  // 4 x 2 wave grid
  const int row16 = lane & 15, hi = lane >> 4;
  const int m0 = blockIdx.x * 64, n0 = blockIdx.y * 64;
  // fast-path vector chunk mapping: 512 x float4 per tile, 2 per thread
  const int cr = t >> 3;  // 0..31 chunk row
  const int cq = t & 7;   // float4 column chunk
  unsigned laA0[2], laA1[2], laB0[2], laB1[2];
#pragma unroll
  for (int b = 0; b < 2; ++b) {
    laA0[b] = lds_off_u32(&sA[b][cr][cq * 4]);
    laA1[b] = lds_off_u32(&sA[b][cr + 32][cq * 4]);
    laB0[b] = lds_off_u32(&sB[b][cr][cq * 4]);
    laB1[b] = lds_off_u32(&sB[b][cr + 32][cq * 4]);
  }
  const float* gA0 = A + (size_t)(m0 + cr) * K + cq * 4;
  const float* gA1 = A + (size_t)(m0 + cr + 32) * K + cq * 4;
  const float* gB0 = W + (size_t)(n0 + cr) * K + cq * 4;
  const float* gB1 = W + (size_t)(n0 + cr + 32) * K + cq * 4;
  f8v acc0 = {}, acc1 = {};

  auto issue_tile = [&](int buf, int kk) {
    async_b128(laA0[buf], gA0 + kk);
    async_b128(laA1[buf], gA1 + kk);
    async_b128(laB0[buf], gB0 + kk);
    async_b128(laB1[buf], gB1 + kk);
  };
  auto fill_tile_scalar = [&](int buf, int kk) {
    const int j = t & 31, i0 = t >> 5;
#pragma unroll
    for (int r = 0; r < 8; ++r) {
      int i = i0 + r * 8;
      int gm = m0 + i, gk = kk + j;
      float va = A[(size_t)(gm < M ? gm : M - 1) * K + (gk < K ? gk : K - 1)];
      sA[buf][i][j] = (gm < M && gk < K) ? va : 0.0f;
      int gn = n0 + i;
      float vb = W[(size_t)(gn < N ? gn : N - 1) * K + (gk < K ? gk : K - 1)];
      sB[buf][i][j] = (gn < N && gk < K) ? vb : 0.0f;
    }
  };
  auto mma_tile = [&](int buf) {
    h16v af = frag_from_lds_f32(&sA[buf][wm * 16 + row16][0], hi);
    h16v bf0 = frag_from_lds_f32(&sB[buf][wn * 32 + row16][0], hi);
    h16v bf1 = frag_from_lds_f32(&sB[buf][wn * 32 + 16 + row16][0], hi);
    acc0 = __builtin_amdgcn_wmma_f32_16x16x32_f16(false, af, false, bf0,
                                                  (short)0, acc0, false, false);
    acc1 = __builtin_amdgcn_wmma_f32_16x16x32_f16(false, af, false, bf1,
                                                  (short)0, acc1, false, false);
  };

  const bool mnFull = (m0 + 64 <= M) && (n0 + 64 <= N);
  if (mnFull) {
    const int K32 = K & ~31;       // full-tile region
    const int nT = K32 >> 5;
    if (nT > 0) {
      issue_tile(0, 0);
      for (int it = 0; it < nT; ++it) {
        if (it + 1 < nT) {
          issue_tile((it + 1) & 1, (it + 1) * 32);  // overlap next DMA
          wait_async4();                            // drain tile `it` only
        } else {
          wait_async0();
        }
        __syncthreads();
        mma_tile(it & 1);
        __syncthreads();
      }
    }
    if (K32 < K) {  // K tail (e.g. K=100 -> 4 columns): scalar clamped fill
      fill_tile_scalar(0, K32);
      __syncthreads();
      mma_tile(0);
      __syncthreads();
    }
  } else {  // edge tiles in M/N: fully scalar clamped path
    for (int kk = 0; kk < K; kk += 32) {
      fill_tile_scalar(0, kk);
      __syncthreads();
      mma_tile(0);
      __syncthreads();
    }
  }

  // C layout: lane n = lane&15; VGPR r -> m = r + 8*(lane>>4)
  const int nA = n0 + wn * 32 + row16;
  const int nB = nA + 16;
#pragma unroll
  for (int r = 0; r < 8; ++r) {
    int m = m0 + wm * 16 + hi * 8 + r;
    if (m < M) {
      if (nA < N) {
        float v = acc0[r] * alpha;
        if (bias) v += bias[nA];
        if (residual) v += residual[(size_t)m * N + nA];
        C[(size_t)m * N + nA] = v;
      }
      if (nB < N) {
        float v = acc1[r] * alpha;
        if (bias) v += bias[nB];
        if (residual) v += residual[(size_t)m * N + nB];
        C[(size_t)m * N + nB] = v;
      }
    }
  }
}

// ---------------------------------------------------------------------------
// C[M,N] += A[Kdim,M]^T @ B[Kdim,N]   (split-K over gridDim.z + f32 atomics)
// Used for ctx[kc,vc] = sum_n K[n,kc] * V[n,vc] with Kdim = #tokens.
// ---------------------------------------------------------------------------
__global__ __launch_bounds__(256) void gemm_atb_atomic(
    const float* __restrict__ A, const float* __restrict__ B,
    float* __restrict__ C, int Kdim, int M, int N) {
  __shared__ __align__(16) float sA[64][36];
  __shared__ __align__(16) float sB[32][36];
  const int t = threadIdx.x;
  const int lane = t & 31, wave = t >> 5;
  const int wm = wave & 3, wn = wave >> 2;
  const int row16 = lane & 15, hi = lane >> 4;
  const int m0 = blockIdx.x * 64, n0 = blockIdx.y * 32;
  const long nsplit = gridDim.z;
  long chunk = ((Kdim + nsplit * 32 - 1) / (nsplit * 32)) * 32;
  long k0 = (long)blockIdx.z * chunk;
  long k1 = k0 + chunk;
  if (k1 > Kdim) k1 = Kdim;
  f8v acc = {};
  for (long kk = k0; kk < k1; kk += 32) {
    {  // sA[m][k] = A[kk+k][m0+m]  (transpose gather, coalesced over m)
      const int i = t & 63, j0 = t >> 6;
#pragma unroll
      for (int r = 0; r < 8; ++r) {
        int j = j0 + r * 4;
        long gk = kk + j;
        int gm = m0 + i;
        float v = A[(size_t)(gk < k1 ? gk : k1 - 1) * M + (gm < M ? gm : M - 1)];
        sA[i][j] = (gk < k1 && gm < M) ? v : 0.0f;
      }
    }
    {  // sB[n][k] = B[kk+k][n0+n]
      const int i = t & 31, j0 = t >> 5;
#pragma unroll
      for (int r = 0; r < 4; ++r) {
        int j = j0 + r * 8;
        long gk = kk + j;
        int gn = n0 + i;
        float v = B[(size_t)(gk < k1 ? gk : k1 - 1) * N + (gn < N ? gn : N - 1)];
        sB[i][j] = (gk < k1 && gn < N) ? v : 0.0f;
      }
    }
    __syncthreads();
    h16v af = frag_from_lds_f32(&sA[wm * 16 + row16][0], hi);
    h16v bf = frag_from_lds_f32(&sB[wn * 16 + row16][0], hi);
    acc = __builtin_amdgcn_wmma_f32_16x16x32_f16(false, af, false, bf,
                                                 (short)0, acc, false, false);
    __syncthreads();
  }
  const int n = n0 + wn * 16 + row16;
#pragma unroll
  for (int r = 0; r < 8; ++r) {
    int m = m0 + wm * 16 + hi * 8 + r;
    if (m < M && n < N) atomicAdd(&C[(size_t)m * N + n], acc[r]);
  }
}

// ---------------- elementwise / reduction kernels ----------------
__global__ void zero_f32(float* p, long n) {
  for (long i = blockIdx.x * (long)blockDim.x + threadIdx.x; i < n;
       i += (long)gridDim.x * blockDim.x)
    p[i] = 0.0f;
}

// Wt[n*K + k] = W[k*N + n]
__global__ void transpose_f32(const float* __restrict__ W, float* __restrict__ Wt,
                              long Kd, long Nd) {
  long total = Kd * Nd;
  for (long i = blockIdx.x * (long)blockDim.x + threadIdx.x; i < total;
       i += (long)gridDim.x * blockDim.x) {
    long k = i / Nd;
    long n = i % Nd;
    Wt[n * Kd + k] = W[i];
  }
}

// patch-embed conv: OIHW weights, token-layout output [n, Cout]
__global__ void conv_pe(const float* __restrict__ x, const float* __restrict__ w,
                        const float* __restrict__ b, float* __restrict__ out,
                        int Hin, int Win, int Hout, int Wout, int ksz,
                        int stride, int pad, int Cin, int Cout) {
  int n = blockIdx.x;
  int c = threadIdx.x;
  if (c >= Cout) return;
  int oy = n / Wout, ox = n % Wout;
  float acc = b[c];
  for (int ci = 0; ci < Cin; ++ci)
    for (int ky = 0; ky < ksz; ++ky) {
      int iy = oy * stride + ky - pad;
      if (iy < 0 || iy >= Hin) continue;
      for (int kx = 0; kx < ksz; ++kx) {
        int ix = ox * stride + kx - pad;
        if (ix < 0 || ix >= Win) continue;
        acc += w[((c * Cin + ci) * ksz + ky) * ksz + kx] *
               x[((size_t)ci * Hin + iy) * Win + ix];
      }
    }
  out[(size_t)n * Cout + c] = acc;
}

// per-column (channel) mean & rstd over M rows of [M,C]  (BatchNorm stats)
__global__ void col_stats(const float* __restrict__ X, float* mean, float* rstd,
                          int M, int C, float eps) {
  int c = blockIdx.x;
  int lane = threadIdx.x & 31, w = threadIdx.x >> 5;
  __shared__ float sh1[8], sh2[8];
  float s = 0.f, s2 = 0.f;
  for (int m = threadIdx.x; m < M; m += blockDim.x) {
    float v = X[(size_t)m * C + c];
    s += v;
    s2 += v * v;
  }
  s = wred_sum(s);
  s2 = wred_sum(s2);
  if (lane == 0) { sh1[w] = s; sh2[w] = s2; }
  __syncthreads();
  if (threadIdx.x == 0) {
    float S = 0.f, S2 = 0.f;
    for (int i = 0; i < 8; ++i) { S += sh1[i]; S2 += sh2[i]; }
    float mu = S / M;
    mean[c] = mu;
    rstd[c] = rsqrtf(S2 / M - mu * mu + eps);
  }
}

__global__ void bn_apply(const float* __restrict__ X, const float* mean,
                         const float* rstd, const float* g, const float* b,
                         float* __restrict__ Y, long total, int C) {
  for (long i = blockIdx.x * (long)blockDim.x + threadIdx.x; i < total;
       i += (long)gridDim.x * blockDim.x) {
    int c = (int)(i % C);
    Y[i] = (X[i] - mean[c]) * rstd[c] * g[c] + b[c];
  }
}

// wave-per-token LayerNorm over C
__global__ void ln_rows(const float* __restrict__ X, const float* g,
                        const float* b, float* __restrict__ Y, int M, int C,
                        float eps) {
  int wave = threadIdx.x >> 5, lane = threadIdx.x & 31;
  int m = blockIdx.x * 8 + wave;
  if (m >= M) return;
  const float* x = X + (size_t)m * C;
  float s = 0.f, s2 = 0.f;
  for (int c = lane; c < C; c += 32) {
    float v = x[c];
    s += v;
    s2 += v * v;
  }
  s = wred_sum(s);
  s2 = wred_sum(s2);
  float mu = s / C, rs = rsqrtf(s2 / C - mu * mu + eps);
  float* y = Y + (size_t)m * C;
  for (int c = lane; c < C; c += 32) y[c] = (x[c] - mu) * rs * g[c] + b[c];
}

// column softmax stats (softmax over tokens, per channel) -- blockDim must be 256
__global__ void softmax_col_stats(const float* __restrict__ X, float* cmax,
                                  float* csum, int M, int C) {
  int c = blockIdx.x;
  int lane = threadIdx.x & 31, w = threadIdx.x >> 5;
  __shared__ float sh[8];
  float mx = -3.0e38f;
  for (int m = threadIdx.x; m < M; m += blockDim.x)
    mx = fmaxf(mx, X[(size_t)m * C + c]);
  mx = wred_max(mx);
  if (lane == 0) sh[w] = mx;
  __syncthreads();
  if (threadIdx.x == 0) {
    float t2 = sh[0];
    for (int i = 1; i < 8; ++i) t2 = fmaxf(t2, sh[i]);
    sh[0] = t2;
  }
  __syncthreads();
  float bmax = sh[0];
  __syncthreads();
  float s = 0.f;
  for (int m = threadIdx.x; m < M; m += blockDim.x)
    s += expf(X[(size_t)m * C + c] - bmax);
  s = wred_sum(s);
  if (lane == 0) sh[w] = s;
  __syncthreads();
  if (threadIdx.x == 0) {
    float t2 = 0.f;
    for (int i = 0; i < 8; ++i) t2 += sh[i];
    cmax[c] = bmax;
    csum[c] = t2;
  }
}

__global__ void softmax_col_apply(const float* __restrict__ X, const float* cmax,
                                  const float* csum, float* __restrict__ Y,
                                  long total, int C) {
  for (long i = blockIdx.x * (long)blockDim.x + threadIdx.x; i < total;
       i += (long)gridDim.x * blockDim.x) {
    int c = (int)(i % C);
    Y[i] = expf(X[i] - cmax[c]) / csum[c];
  }
}

// wave-per-row softmax over C
__global__ void softmax_rows(const float* __restrict__ X, float* __restrict__ Y,
                             int M, int C) {
  int wave = threadIdx.x >> 5, lane = threadIdx.x & 31;
  int m = blockIdx.x * 8 + wave;
  if (m >= M) return;
  const float* x = X + (size_t)m * C;
  float mx = -3.0e38f;
  for (int c = lane; c < C; c += 32) mx = fmaxf(mx, x[c]);
  mx = wred_max(mx);
  float s = 0.f;
  for (int c = lane; c < C; c += 32) s += expf(x[c] - mx);
  s = wred_sum(s);
  float inv = 1.0f / s;
  float* y = Y + (size_t)m * C;
  for (int c = lane; c < C; c += 32) y[c] = expf(x[c] - mx) * inv;
}

__global__ void row_mean(const float* __restrict__ X, float* __restrict__ Y,
                         int M, int C) {
  int wave = threadIdx.x >> 5, lane = threadIdx.x & 31;
  int m = blockIdx.x * 8 + wave;
  if (m >= M) return;
  const float* x = X + (size_t)m * C;
  float s = 0.f;
  for (int c = lane; c < C; c += 32) s += x[c];
  s = wred_sum(s);
  if (lane == 0) Y[m] = s / C;
}

// depthwise 3x3 conv, token layout [n, E]
__global__ void dwconv3(const float* __restrict__ f1, const float* __restrict__ w,
                        const float* __restrict__ b, float* __restrict__ out,
                        int H, int W, int E) {
  int n = blockIdx.x;
  int e = blockIdx.y * blockDim.x + threadIdx.x;
  if (e >= E) return;
  int y = n / W, x = n % W;
  float acc = b[e];
#pragma unroll
  for (int ky = 0; ky < 3; ++ky) {
    int iy = y + ky - 1;
    if (iy < 0 || iy >= H) continue;
#pragma unroll
    for (int kx = 0; kx < 3; ++kx) {
      int ix = x + kx - 1;
      if (ix < 0 || ix >= W) continue;
      acc += w[e * 9 + ky * 3 + kx] * f1[((size_t)iy * W + ix) * E + e];
    }
  }
  out[(size_t)n * E + e] = acc;
}

// ax = gelu(LN(f1 + dw)); out may alias dw (wave owns its row)
__global__ void add_ln_gelu(const float* __restrict__ f1,
                            const float* __restrict__ dw, const float* g,
                            const float* b, float* __restrict__ out, int M,
                            int C, float eps) {
  int wave = threadIdx.x >> 5, lane = threadIdx.x & 31;
  int m = blockIdx.x * 8 + wave;
  if (m >= M) return;
  const float* a = f1 + (size_t)m * C;
  const float* d = dw + (size_t)m * C;
  float s = 0.f, s2 = 0.f;
  for (int c = lane; c < C; c += 32) {
    float v = a[c] + d[c];
    s += v;
    s2 += v * v;
  }
  s = wred_sum(s);
  s2 = wred_sum(s2);
  float mu = s / C, rs = rsqrtf(s2 / C - mu * mu + eps);
  float* o = out + (size_t)m * C;
  for (int c = lane; c < C; c += 32) {
    float v = a[c] + d[c];
    v = (v - mu) * rs * g[c] + b[c];
    o[c] = 0.5f * v * (1.0f + erff(v * 0.70710678118f));
  }
}

// bilinear upsample token layout [Hs*Ws, C] -> strided dst rows (writes xcat cols)
__global__ void upsample_bilinear(const float* __restrict__ src,
                                  float* __restrict__ dst, int Hs, int Ws,
                                  int Ht, int Wt, int C, int dstStride,
                                  int dstOff) {
  int n = blockIdx.x;
  int c = blockIdx.y * blockDim.x + threadIdx.x;
  if (c >= C) return;
  int y = n / Wt, x = n % Wt;
  float sy = (y + 0.5f) * (float)Hs / (float)Ht - 0.5f;
  float sx = (x + 0.5f) * (float)Ws / (float)Wt - 0.5f;
  float fy0 = floorf(sy), fx0 = floorf(sx);
  float ay = sy - fy0, ax = sx - fx0;
  int y0 = (int)fy0, x0 = (int)fx0;
  int y1 = y0 + 1, x1 = x0 + 1;
  y0 = y0 < 0 ? 0 : (y0 >= Hs ? Hs - 1 : y0);
  y1 = y1 < 0 ? 0 : (y1 >= Hs ? Hs - 1 : y1);
  x0 = x0 < 0 ? 0 : (x0 >= Ws ? Ws - 1 : x0);
  x1 = x1 < 0 ? 0 : (x1 >= Ws ? Ws - 1 : x1);
  float v00 = src[((size_t)y0 * Ws + x0) * C + c];
  float v01 = src[((size_t)y0 * Ws + x1) * C + c];
  float v10 = src[((size_t)y1 * Ws + x0) * C + c];
  float v11 = src[((size_t)y1 * Ws + x1) * C + c];
  float v = v00 * (1 - ay) * (1 - ax) + v01 * (1 - ay) * ax +
            v10 * ay * (1 - ax) + v11 * ay * ax;
  dst[(size_t)n * dstStride + dstOff + c] = v;
}

__global__ void copy_cols(const float* __restrict__ src, float* __restrict__ dst,
                          long M, int Cs, int dstStride, int off) {
  long total = M * Cs;
  for (long i = blockIdx.x * (long)blockDim.x + threadIdx.x; i < total;
       i += (long)gridDim.x * blockDim.x) {
    long n = i / Cs;
    int c = (int)(i % Cs);
    dst[(size_t)n * dstStride + off + c] = src[i];
  }
}

// raw out -> image layout d_out + bn3-applied out2 (token layout)
__global__ void out_transpose_bn(const float* __restrict__ outTok,
                                 const float* mean, const float* rstd,
                                 const float* g, const float* b,
                                 float* __restrict__ outImg,
                                 float* __restrict__ out2Tok, long Nn, int C) {
  long total = Nn * C;
  for (long i = blockIdx.x * (long)blockDim.x + threadIdx.x; i < total;
       i += (long)gridDim.x * blockDim.x) {
    long n = i / C;
    int c = (int)(i % C);
    float v = outTok[i];
    outImg[(size_t)c * Nn + n] = v;
    out2Tok[i] = (v - mean[c]) * rstd[c] * g[c] + b[c];
  }
}

// patch flatten: P[m, c*1024 + py*32 + px] = out2[(y,x) shifted] ; m = ph*8+pw
__global__ void gather_patches(const float* __restrict__ out2Tok,
                               float* __restrict__ P, int off, int HW, int C) {
  long total = 64L * C * 1024;
  for (long i = blockIdx.x * (long)blockDim.x + threadIdx.x; i < total;
       i += (long)gridDim.x * blockDim.x) {
    long m = i / ((long)C * 1024);
    long k = i % ((long)C * 1024);
    int c = (int)(k / 1024);
    int py = (int)((k % 1024) / 32);
    int px = (int)(k % 32);
    int ph = (int)(m / 8), pw = (int)(m % 8);
    int y = ph * 32 + py + off, x = pw * 32 + px + off;
    P[i] = out2Tok[((size_t)y * HW + x) * C + c];
  }
}

// ---------------------------------------------------------------------------
extern "C" void kernel_launch(void* const* d_in, const int* in_sizes, int n_in,
                              void* d_out, int out_size, void* d_ws,
                              size_t ws_size, hipStream_t stream) {
  (void)in_sizes; (void)n_in; (void)out_size; (void)ws_size;
  auto P = [&](int i) { return (const float*)d_in[i]; };
  const float* X = P(0);
  // param indices (setup_inputs insertion order, depth-first):
  // 1..4 pe_f{w,b,g,beta}  5..8 pe_s  9..12 pe_l
  // 13.. blk_f(22)  35.. blk_s  57.. blk_l
  // 79 lin_up  80 lin_down  81 bn3_g  82 bn3_b  83 pp_w  84 pp_b

  // ---- outputs ----
  float* o_out = (float*)d_out;              // 100*257*257
  float* o_attn = o_out + (size_t)DD * NFT;  // 64*64
  float* o_amf = o_attn + 64 * 64;           // 257*257
  float* o_ams = o_amf + NFT;                // 257*257
  float* o_aml = o_ams + NFT;                // 129*129
  float* o_ctxf = o_aml + NLT;               // 100*100
  float* o_ctxs = o_ctxf + DD * DD;
  float* o_ctxl = o_ctxs + DD * DD;

  // ---- workspace carve ----
  float* wp = (float*)d_ws;
  size_t off = 0;
  auto alloc = [&](size_t n) { float* p = wp + off; off += (n + 31) & ~(size_t)31; return p; };
  float* bufXF = alloc((size_t)NFT * DD);
  float* bufXS = alloc((size_t)NFT * DD);
  float* bufXL = alloc((size_t)NLT * DD);
  float* wsA = alloc((size_t)NFT * DD);
  float* wsQ = alloc((size_t)NFT * DD);
  float* wsK = alloc((size_t)NFT * DD);
  float* wsV = alloc((size_t)NFT * DD);
  float* wsB = alloc((size_t)NFT * DD);
  float* wsC = alloc((size_t)NFT * DD);
  float* wsF1 = alloc((size_t)NFT * EE);
  float* wsF2 = alloc((size_t)NFT * EE);
  float* xcat = alloc((size_t)NFT * 600);
  float* wTa = alloc((size_t)102400 * DD);  // transposed-weight scratch
  float* stat0 = alloc(1024);
  float* stat1 = alloc(1024);
  // post-block reuse
  float* outTok = wsQ;             // [NFT,100]
  float* out2Tok = wsK;            // [NFT,100]
  float* Qp = wsF1;                // [64,102400]
  float* Kp = wsF2;                // [64,102400]
  float* Qm = wsV;                 // [64,100]
  float* Km = wsV + 8192;          // [64,100]
  float* scoresBuf = wsV + 16384;  // [64,64]

  auto gemm = [&](const float* A, const float* Wt, const float* bias,
                  const float* res, float* C, int M, int K, int N,
                  float alpha) {
    dim3 g((M + 63) / 64, (N + 63) / 64);
    gemm_wmma_f16<<<g, 256, 0, stream>>>(A, Wt, bias, res, C, M, K, N, alpha);
  };
  auto ew_grid = [](long total) {
    long b = (total + 255) / 256;
    if (b > 16384) b = 16384;
    return (int)b;
  };
  auto transpose = [&](const float* W, float* Wt, long Kd, long Nd) {
    transpose_f32<<<ew_grid(Kd * Nd), 256, 0, stream>>>(W, Wt, Kd, Nd);
  };

  // ---- patch embeds: conv + BN (column stats over tokens) ----
  auto run_pe = [&](int pb, int ksz, int stride, int pad, int Hout, int Wout,
                    float* dst) {
    int N = Hout * Wout;
    conv_pe<<<N, 128, 0, stream>>>(X, P(pb + 0), P(pb + 1), dst, IMGX, IMGX,
                                   Hout, Wout, ksz, stride, pad, 3, DD);
    col_stats<<<DD, 256, 0, stream>>>(dst, stat0, stat1, N, DD, 1e-5f);
    bn_apply<<<ew_grid((long)N * DD), 256, 0, stream>>>(
        dst, stat0, stat1, P(pb + 2), P(pb + 3), dst, (long)N * DD, DD);
  };
  run_pe(1, 1, 1, 0, 257, 257, bufXF);
  run_pe(5, 3, 1, 1, 257, 257, bufXS);
  run_pe(9, 3, 2, 1, 129, 129, bufXL);

  // ---- transformer block (token layout [N, 100]) ----
  auto run_block = [&](int bp, float* tok, int H, int W, float* outBuf,
                       float* ctxOut, float* amapOut) {
    int N = H * W;
    int rb8 = (N + 7) / 8;
    // n1 = LN1(tok)
    ln_rows<<<rb8, 256, 0, stream>>>(tok, P(bp + 0), P(bp + 1), wsA, N, DD, 1e-5f);
    // q,k,v projections: out = n1 @ W^T + b  (qw/kw/vw already [N,K])
    gemm(wsA, P(bp + 2), P(bp + 3), nullptr, wsQ, N, DD, DD, 1.f);
    gemm(wsA, P(bp + 4), P(bp + 5), nullptr, wsK, N, DD, DD, 1.f);
    gemm(wsA, P(bp + 6), P(bp + 7), nullptr, wsV, N, DD, DD, 1.f);
    // k: softmax over tokens (per channel)
    softmax_col_stats<<<DD, 256, 0, stream>>>(wsK, stat0, stat1, N, DD);
    softmax_col_apply<<<ew_grid((long)N * DD), 256, 0, stream>>>(
        wsK, stat0, stat1, wsK, (long)N * DD, DD);
    // amap = mean over channels of softmaxed k
    row_mean<<<rb8, 256, 0, stream>>>(wsK, amapOut, N, DD);
    // q: softmax over channels (per token)
    softmax_rows<<<rb8, 256, 0, stream>>>(wsQ, wsQ, N, DD);
    // ctx[kc,vc] = sum_n K[n,kc]*V[n,vc]  (split-K WMMA + atomics)
    zero_f32<<<40, 256, 0, stream>>>(ctxOut, (long)DD * DD);
    {
      dim3 g((DD + 63) / 64, (DD + 31) / 32, 32);
      gemm_atb_atomic<<<g, 256, 0, stream>>>(wsK, wsV, ctxOut, N, DD, DD);
    }
    // att = q @ ctx  (ctx is [K,N] -> transpose to [N,K] first)
    transpose(ctxOut, wTa, DD, DD);
    gemm(wsQ, wTa, nullptr, nullptr, wsB, N, DD, DD, 1.f);
    // r-projection with residual tok -> tx  (rw already [N,K])
    gemm(wsB, P(bp + 8), P(bp + 9), tok, wsC, N, DD, DD, 1.f);
    // FFN: n2 = LN2(tx); f1 = n2 @ fc1_w + b; dw conv; ax = gelu(LN(f1+dw))
    ln_rows<<<rb8, 256, 0, stream>>>(wsC, P(bp + 10), P(bp + 11), wsA, N, DD, 1e-5f);
    transpose(P(bp + 12), wTa, DD, EE);  // fc1_w [100,400] -> [400,100]
    gemm(wsA, wTa, P(bp + 13), nullptr, wsF1, N, DD, EE, 1.f);
    {
      dim3 g(N, (EE + 255) / 256);
      dwconv3<<<g, 256, 0, stream>>>(wsF1, P(bp + 14), P(bp + 15), wsF2, H, W, EE);
    }
    add_ln_gelu<<<rb8, 256, 0, stream>>>(wsF1, wsF2, P(bp + 16), P(bp + 17),
                                         wsF2, N, EE, 1e-5f);
    // mx = tx + ax @ fc2_w + b ; bn2 -> block output
    transpose(P(bp + 18), wTa, EE, DD);  // fc2_w [400,100] -> [100,400]
    gemm(wsF2, wTa, P(bp + 19), wsC, wsB, N, EE, DD, 1.f);
    col_stats<<<DD, 256, 0, stream>>>(wsB, stat0, stat1, N, DD, 1e-5f);
    bn_apply<<<ew_grid((long)N * DD), 256, 0, stream>>>(
        wsB, stat0, stat1, P(bp + 20), P(bp + 21), outBuf, (long)N * DD, DD);
  };
  run_block(13, bufXF, 257, 257, bufXF, o_ctxf, o_amf);
  run_block(35, bufXS, 257, 257, bufXS, o_ctxs, o_ams);
  run_block(57, bufXL, 129, 129, bufXL, o_ctxl, o_aml);

  // ---- fuse scales: lin_up -> bilinear upsample -> concat -> lin_down ----
  transpose(P(79), wTa, DD, 4 * DD);  // lin_up [100,400] -> [400,100]
  gemm(bufXL, wTa, nullptr, nullptr, wsF1, NLT, DD, 4 * DD, 1.f);
  {
    dim3 g(NFT, (4 * DD + 255) / 256);
    upsample_bilinear<<<g, 256, 0, stream>>>(wsF1, xcat, 129, 129, 257, 257,
                                             4 * DD, 600, 0);
  }
  copy_cols<<<ew_grid((long)NFT * DD), 256, 0, stream>>>(bufXS, xcat, NFT, DD, 600, 400);
  copy_cols<<<ew_grid((long)NFT * DD), 256, 0, stream>>>(bufXF, xcat, NFT, DD, 600, 500);
  transpose(P(80), wTa, 600, DD);  // lin_down [600,100] -> [100,600]
  gemm(xcat, wTa, nullptr, nullptr, outTok, NFT, 600, DD, 1.f);

  // ---- bn3 + emit raw 'out' (NCHW) + out2 ----
  col_stats<<<DD, 256, 0, stream>>>(outTok, stat0, stat1, NFT, DD, 1e-5f);
  out_transpose_bn<<<ew_grid((long)NFT * DD), 256, 0, stream>>>(
      outTok, stat0, stat1, P(81), P(82), o_out, out2Tok, NFT, DD);

  // ---- patch attention head ----
  gather_patches<<<4096, 256, 0, stream>>>(out2Tok, Qp, 1, 257, DD);
  gather_patches<<<4096, 256, 0, stream>>>(out2Tok, Kp, 0, 257, DD);
  transpose(P(83), wTa, 102400, DD);  // pp_w [102400,100] -> [100,102400]
  gemm(Qp, wTa, P(84), nullptr, Qm, 64, DD * 32 * 32, DD, 1.f);
  gemm(Kp, wTa, P(84), nullptr, Km, 64, DD * 32 * 32, DD, 1.f);
  gemm(Qm, Km, nullptr, nullptr, scoresBuf, 64, DD, 64, 0.17677669529f);
  softmax_rows<<<(64 + 7) / 8, 256, 0, stream>>>(scoresBuf, o_attn, 64, 64);
}
